// SNNlifadex_4801773437473
// MI455X (gfx1250) — compile-verified, matrix operand order checked by reading, not run on hardware
//
#include <hip/hip_runtime.h>
#include <hip/hip_bf16.h>
#include <stdint.h>

#define TT 1000
#define BB 64
#define FF 512
#define HH 1024
#define OO 256
#define BH (BB * HH)

// fused constants (DT = 1e-3)
#define K_DT_TAUMEM 0.1f            // DT * TAU_MEM_INV
#define K_DT_TAUSYN 0.2f            // DT * TAU_SYN_INV
#define K_DT_TAUADA 0.002f          // DT * TAU_ADA_INV
#define K_DT_TAUFLT 0.2231435511314f
#define K_VTH    1.0f
#define K_DELTA  0.5f
#define K_ADA_A  4.0f
#define K_ADA_B  0.02f

typedef __attribute__((ext_vector_type(16))) __bf16 v16bf_t;
typedef __attribute__((ext_vector_type(8)))  __bf16 v8bf_t;
typedef __attribute__((ext_vector_type(8)))  float  v8f_t;

__device__ __forceinline__ v8f_t vzero8() {
    v8f_t z = {0.f, 0.f, 0.f, 0.f, 0.f, 0.f, 0.f, 0.f};
    return z;
}

// ---------------------------------------------------------------------------
// WMMA fragment helpers. ISA 16-bit A-matrix 16x32 lane layout (wave32):
//   lane l: row M = l&15 ; kb = (l<16) ? 0 : 8
//   halves 0..7  = K kb+0 .. kb+7   (contiguous)
//   halves 8..15 = K kb+16 .. kb+23 (contiguous)
// ---------------------------------------------------------------------------
__device__ __forceinline__ v16bf_t frag_from_bf16(const __bf16* p) {
    v8bf_t lo = *(const v8bf_t*)(p);
    v8bf_t hi = *(const v8bf_t*)(p + 16);
    v16bf_t r;
#pragma unroll
    for (int i = 0; i < 8; ++i) { r[i] = lo[i]; r[8 + i] = hi[i]; }
    return r;
}

__device__ __forceinline__ v8f_t wmma_bf16(v16bf_t a, v16bf_t b, v8f_t c) {
#if defined(__AMDGCN__)
    return __builtin_amdgcn_wmma_f32_16x16x32_bf16(false, a, false, b, (short)0, c,
                                                   false, false);
#else
    (void)a; (void)b;
    return c;
#endif
}

// ---------------------------------------------------------------------------
// Async global -> LDS staging (gfx1250 ASYNCcnt path).
// Generic pointers to __shared__ carry the LDS byte offset in their low 32
// bits (ISA 10.2), which is what VDST wants.
// ---------------------------------------------------------------------------
__device__ __forceinline__ void async_copy16(uint32_t lds_byte_off, const void* gsrc) {
#if defined(__AMDGCN__)
    asm volatile("global_load_async_to_lds_b128 %0, %1, off"
                 :: "v"(lds_byte_off), "v"(gsrc)
                 : "memory");
#else
    (void)lds_byte_off; (void)gsrc;
#endif
}

__device__ __forceinline__ void wait_async_all() {
#if defined(__AMDGCN__)
    asm volatile("s_wait_asynccnt 0" ::: "memory");
#endif
}

// Copy `bytes` (multiple of 4096) contiguously from global to LDS, 16B/lane.
__device__ __forceinline__ void stage_to_lds(void* smem_dst, const void* gsrc,
                                             int bytes, int tid) {
    uint32_t dbase = (uint32_t)(uintptr_t)smem_dst + (uint32_t)tid * 16u;
    const char* s = (const char*)gsrc + (size_t)tid * 16;
    for (int off = 0; off < bytes; off += 256 * 16) {
        async_copy16(dbase + (uint32_t)off, s + off);
    }
    wait_async_all();
}

// ---------------------------------------------------------------------------
// K0: f32 -> bf16 conversion
// ---------------------------------------------------------------------------
__global__ void cvt_bf16_kernel(const float* __restrict__ s, __bf16* __restrict__ d, int n) {
    int i = blockIdx.x * blockDim.x + threadIdx.x;
    if (i < n) d[i] = (__bf16)s[i];
}

// ---------------------------------------------------------------------------
// Generic bf16 WMMA GEMM: C[m,n] = sum_k A[m,k] * Bw[n,k] (+ bias[n])
// A: [M,K] bf16 row-major, Bw: [N,K] bf16 row-major, C: [M,N] f32.
// grid = (M/16, N/256), block = 256 (8 waves). Per block: one 16-row A tile
// staged to LDS via async loads; each wave computes a 16x32 strip with two
// independent accumulators.
// ---------------------------------------------------------------------------
__global__ void __launch_bounds__(256)
gemm_bf16_kernel(const __bf16* __restrict__ A, const __bf16* __restrict__ Bw,
                 const float* __restrict__ bias, float* __restrict__ C,
                 int K, int N) {
    __shared__ __bf16 smA[16 * 1024];  // up to K=1024 (32 KB)
    const int tid  = threadIdx.x;
    const int lane = tid & 31;
    const int wave = tid >> 5;
    const int m0 = blockIdx.x << 4;
    const int n0 = (blockIdx.y << 8) + (wave << 5);
    const int kb  = (lane < 16) ? 0 : 8;
    const int l15 = lane & 15;

    stage_to_lds(smA, A + (size_t)m0 * K, 16 * K * 2, tid);
    __syncthreads();

    const __bf16* ar  = smA + l15 * K + kb;                      // LDS (ds_load)
    const __bf16* bp0 = Bw + (size_t)(n0 + l15) * K + kb;        // global
    const __bf16* bp1 = Bw + (size_t)(n0 + 16 + l15) * K + kb;

    v8f_t c0 = vzero8(), c1 = vzero8();
#pragma unroll 4
    for (int k = 0; k < K; k += 32) {
        v16bf_t a  = frag_from_bf16(ar + k);
        v16bf_t b0 = frag_from_bf16(bp0 + k);
        v16bf_t b1 = frag_from_bf16(bp1 + k);
        c0 = wmma_bf16(a, b0, c0);
        c1 = wmma_bf16(a, b1, c1);
    }

    const int n = n0 + l15;
    const float bi0 = bias ? bias[n] : 0.0f;
    const float bi1 = bias ? bias[n + 16] : 0.0f;
    float* orow = C + (size_t)(m0 + ((lane >> 4) << 3)) * N + n;
#pragma unroll
    for (int v = 0; v < 8; ++v) {
        orow[(size_t)v * N]      = c0[v] + bi0;
        orow[(size_t)v * N + 16] = c1[v] + bi1;
    }
}

// ---------------------------------------------------------------------------
// One recurrent timestep: rcur = z_prev @ w_rec^T (WMMA, z_prev staged to LDS
// via async loads) fused with the adaptive-exponential LIF update.
// grid = (H/256, B/16), block = 256.
// ---------------------------------------------------------------------------
__global__ void __launch_bounds__(256)
rec_step_kernel(const __bf16* __restrict__ zprev, const __bf16* __restrict__ wrec,
                const float* __restrict__ xin_t,
                float* __restrict__ vS, float* __restrict__ iS, float* __restrict__ aS,
                __bf16* __restrict__ zout, int t, float* __restrict__ fin) {
    __shared__ __bf16 smZ[16 * HH];  // 32 KB: z_prev rows m0..m0+15
    const int tid  = threadIdx.x;
    const int lane = tid & 31;
    const int wave = tid >> 5;
    const int m0 = blockIdx.y << 4;                 // batch tile
    const int n0 = (blockIdx.x << 8) + (wave << 5); // hidden strip (32 wide)
    const int kb  = (lane < 16) ? 0 : 8;
    const int l15 = lane & 15;

    v8f_t acc[2];
    acc[0] = vzero8();
    acc[1] = vzero8();

    if (t > 0) {
        stage_to_lds(smZ, zprev + (size_t)m0 * HH, 16 * HH * 2, tid);
        __syncthreads();
        const __bf16* ar  = smZ + l15 * HH + kb;
        const __bf16* bp0 = wrec + (size_t)(n0 + l15) * HH + kb;
        const __bf16* bp1 = wrec + (size_t)(n0 + 16 + l15) * HH + kb;
#pragma unroll 4
        for (int k = 0; k < HH; k += 32) {
            v16bf_t a = frag_from_bf16(ar + k);
            acc[0] = wmma_bf16(a, frag_from_bf16(bp0 + k), acc[0]);
            acc[1] = wmma_bf16(a, frag_from_bf16(bp1 + k), acc[1]);
        }
    }

    const int mbase = m0 + ((lane >> 4) << 3);
#pragma unroll
    for (int s = 0; s < 2; ++s) {
        const int h = n0 + s * 16 + l15;
#pragma unroll
        for (int v = 0; v < 8; ++v) {
            const int idx = (mbase + v) * HH + h;
            float vv = vS[idx];
            float ii = iS[idx];
            float aa = aS[idx];
            // V_LEAK = 0, V_RESET = 0
            float dv    = K_DT_TAUMEM * (-vv + K_DELTA * __expf((vv - K_VTH) * 2.0f) + ii - aa);
            float v_dec = vv + dv;
            float i_dec = ii - K_DT_TAUSYN * ii;
            float a_dec = aa + K_DT_TAUADA * (K_ADA_A * vv - aa);
            float z     = (v_dec - K_VTH) > 0.0f ? 1.0f : 0.0f;
            float v_new = (1.0f - z) * v_dec;
            float i_new = i_dec + xin_t[idx] + acc[s][v];
            float a_new = a_dec + z * K_ADA_B;
            vS[idx] = v_new;
            iS[idx] = i_new;
            aS[idx] = a_new;
            zout[idx] = (__bf16)z;
            if (fin) {  // t == T-1: state_final = (z, v, i, a)
                fin[idx]          = z;
                fin[BH + idx]     = v_new;
                fin[2 * BH + idx] = i_new;
                fin[3 * BH + idx] = a_new;
            }
        }
    }
}

// ---------------------------------------------------------------------------
// Exponential filter scan over T: out[0] = lin[0]; y += k*(lin[t]-y).
// One thread per (b,o); fully coalesced per timestep.
// ---------------------------------------------------------------------------
__global__ void filter_kernel(const float* __restrict__ lin, float* __restrict__ out) {
    const int idx = blockIdx.x * blockDim.x + threadIdx.x;  // < B*O
    float y = lin[idx];
    out[idx] = y;
    for (int t = 1; t < TT; ++t) {
        const size_t o = (size_t)t * (BB * OO) + idx;
        y += K_DT_TAUFLT * (lin[o] - y);
        out[o] = y;
    }
}

// ---------------------------------------------------------------------------
extern "C" void kernel_launch(void* const* d_in, const int* in_sizes, int n_in,
                              void* d_out, int out_size, void* d_ws, size_t ws_size,
                              hipStream_t stream) {
    (void)in_sizes; (void)n_in; (void)out_size; (void)ws_size;
    const float* x     = (const float*)d_in[0];
    const float* w_in  = (const float*)d_in[1];
    const float* w_rec = (const float*)d_in[2];
    const float* w_out = (const float*)d_in[3];
    const float* b_out = (const float*)d_in[4];
    float* out = (float*)d_out;

    char* base = (char*)d_ws;
    size_t off = 0;
    auto take = [&](size_t bytes) -> void* {
        void* r = base + off;
        off += (bytes + 255) & ~(size_t)255;
        return r;
    };
    __bf16* w_in_b  = (__bf16*)take((size_t)HH * FF * 2);
    __bf16* w_rec_b = (__bf16*)take((size_t)HH * HH * 2);
    __bf16* w_out_b = (__bf16*)take((size_t)OO * HH * 2);
    __bf16* z0      = (__bf16*)take((size_t)BH * 2);      // zeroed: z at t=-1
    float*  vS      = (float*)take((size_t)BH * 4);
    float*  iS      = (float*)take((size_t)BH * 4);
    float*  aS      = (float*)take((size_t)BH * 4);
    __bf16* x_b     = (__bf16*)take((size_t)TT * BB * FF * 2);
    float*  xin     = (float*)take((size_t)TT * BH * 4);
    __bf16* spikes  = (__bf16*)take((size_t)TT * BH * 2);
    float*  lin     = (float*)take((size_t)TT * BB * OO * 4);

    // zero z0 + v/i/a (contiguous region) every call — deterministic state
    size_t zbytes = (size_t)((char*)x_b - (char*)z0);
    hipMemsetAsync(z0, 0, zbytes, stream);

    cvt_bf16_kernel<<<(HH * FF + 255) / 256, 256, 0, stream>>>(w_in, w_in_b, HH * FF);
    cvt_bf16_kernel<<<(HH * HH + 255) / 256, 256, 0, stream>>>(w_rec, w_rec_b, HH * HH);
    cvt_bf16_kernel<<<(OO * HH + 255) / 256, 256, 0, stream>>>(w_out, w_out_b, OO * HH);
    cvt_bf16_kernel<<<(TT * BB * FF + 255) / 256, 256, 0, stream>>>(x, x_b, TT * BB * FF);

    // input projection for all timesteps: one big WMMA GEMM
    gemm_bf16_kernel<<<dim3((TT * BB) / 16, HH / 256), 256, 0, stream>>>(
        x_b, w_in_b, nullptr, xin, FF, HH);

    // sequential recurrence: 1000 fused GEMM + LIF-update kernels
    for (int t = 0; t < TT; ++t) {
        const __bf16* zp = (t == 0) ? z0 : (spikes + (size_t)(t - 1) * BH);
        float* fin = (t == TT - 1) ? (out + (size_t)TT * BB * OO) : nullptr;
        rec_step_kernel<<<dim3(HH / 256, BB / 16), 256, 0, stream>>>(
            zp, w_rec_b, xin + (size_t)t * BH, vS, iS, aS,
            spikes + (size_t)t * BH, t, fin);
    }

    // readout GEMM + bias
    gemm_bf16_kernel<<<dim3((TT * BB) / 16, OO / 256), 256, 0, stream>>>(
        spikes, w_out_b, b_out, lin, HH, OO);

    // exponential low-pass filter over time
    filter_kernel<<<(BB * OO) / 256, 256, 0, stream>>>(lin, out);
}
